// AgentPredictionHead_35828617183421
// MI455X (gfx1250) — matrix operand order repaired and checked by reading, not compiled, not run on hardware
//
#include <hip/hip_runtime.h>

typedef unsigned short u16;
typedef __attribute__((ext_vector_type(16))) __bf16 bf16x16;
typedef __attribute__((ext_vector_type(8)))  __bf16 bf16x8;
typedef __attribute__((ext_vector_type(8)))  float  f32x8;

#define D_    512
#define H_    128
#define T_    10
#define ROWS  128
#define HP    136   // LDS row pitch in u16 elements (128 + 8 pad -> no 64-bank aliasing)

// small-vector LDS layout (float offsets)
#define SV_LSTM   0     // 512
#define SV_XB     512   // 128
#define SV_B1     640   // 128
#define SV_B2     768   // 128
#define SV_BAB    896   // 128 (ba+btb)
#define SV_WFC    1024  // 256
#define SV_BFC    1280  // 2
#define SV_BINIT  1284  // 256
#define SV_FLOATS 1540

__device__ __forceinline__ u16 f2bf(float f) {
  unsigned u = __float_as_uint(f);
  u += 0x7FFFu + ((u >> 16) & 1u);          // round-to-nearest-even
  return (u16)(u >> 16);
}
__device__ __forceinline__ float bf2f(u16 s) {
  return __uint_as_float(((unsigned)s) << 16);
}
__device__ __forceinline__ float sigm(float x) {
  x = fminf(fmaxf(x, -30.0f), 30.0f);
  return 1.0f / (1.0f + __expf(-x));
}
__device__ __forceinline__ float tanh_fast(float x) {
  x = fminf(fmaxf(x, -15.0f), 15.0f);
  float e = __expf(-2.0f * x);
  return (1.0f - e) / (1.0f + e);
}
__device__ __forceinline__ f32x8 splat8(float v) {
  f32x8 r = {v, v, v, v, v, v, v, v};
  return r;
}

// A fragment (16x32 bf16, M x K) from LDS row-major [row][k] buffer.
// Lane l holds row mBase+(l&15); lo lanes K {0..7,16..23}, hi lanes K {8..15,24..31}.
__device__ __forceinline__ bf16x16 ldsA(const u16* buf, int mBase, int kBase, int lane) {
  int m  = mBase + (lane & 15);
  int hv = (lane >> 4) << 3;                 // 0 or 8
  const bf16x8* p0 = (const bf16x8*)(buf + m * HP + kBase + hv);
  const bf16x8* p1 = (const bf16x8*)(buf + m * HP + kBase + 16 + hv);
  bf16x8 lo = *p0, hi = *p1;
  return __builtin_shufflevector(lo, hi, 0,1,2,3,4,5,6,7,8,9,10,11,12,13,14,15);
}

// B fragment (32x16 bf16, K x N) from global weights stored [n][k] (k contiguous).
// Lane l = column nBase+(l&15); lo lanes K {0..7,16..23}, hi lanes K {8..15,24..31}.
__device__ __forceinline__ bf16x16 gB(const u16* W, int pitchK, int nBase, int kBase, int lane) {
  int n = nBase + (lane & 15);
  int k = kBase + ((lane >> 4) << 3);
  bf16x8 lo = *(const bf16x8*)(W + (size_t)n * pitchK + k);
  bf16x8 hi = *(const bf16x8*)(W + (size_t)n * pitchK + k + 16);
  return __builtin_shufflevector(lo, hi, 0,1,2,3,4,5,6,7,8,9,10,11,12,13,14,15);
}

__device__ __forceinline__ f32x8 wmma_bf16(bf16x16 a, bf16x16 b, f32x8 c) {
  return __builtin_amdgcn_wmma_f32_16x16x32_bf16(false, a, false, b, (short)0, c, false, false);
}

// Store a C/D fragment (activated) into LDS bf16 row-major buffer.
// VGPR e <-> row mBase + e + 8*(lane>=16); col nBase + (lane&15).
__device__ __forceinline__ void stC(u16* buf, int mBase, int nBase, int lane, f32x8 v) {
  int col = nBase + (lane & 15);
  int rb  = mBase + ((lane >> 4) << 3);
#pragma unroll
  for (int e = 0; e < 8; ++e)
    buf[(rb + e) * HP + col] = f2bf(v[e]);
}

// ---------------------------------------------------------------------------
// Prep: rank-1 input projections + f32 -> bf16 weight conversion (B layout =
// original (out,in) row-major, i.e. column-major for the B operand).
// ---------------------------------------------------------------------------
__global__ __launch_bounds__(256) void cfc_prep(
    const float* __restrict__ ctrl,  const float* __restrict__ Winit,
    const float* __restrict__ Wi,    const float* __restrict__ bi,
    const float* __restrict__ Wr,    const float* __restrict__ Wb,
    const float* __restrict__ bb,    const float* __restrict__ W1,
    const float* __restrict__ W2,    const float* __restrict__ Wa,
    const float* __restrict__ Wtb,
    u16* __restrict__ WinitB, u16* __restrict__ WrB, u16* __restrict__ WbhB,
    u16* __restrict__ W1B, u16* __restrict__ W2B, u16* __restrict__ WaB,
    u16* __restrict__ WtbB, float* __restrict__ lstm_in, float* __restrict__ xb_in)
{
  int tid = blockIdx.x * blockDim.x + threadIdx.x;
  int nT  = gridDim.x * blockDim.x;

  for (int j = tid; j < 4 * H_; j += nT) {          // lstm_in = Wi @ ctrl + bi
    float a = bi[j];
    for (int k = 0; k < D_; ++k) a = fmaf(Wi[(size_t)j * D_ + k], ctrl[k], a);
    lstm_in[j] = a;
  }
  for (int j = tid; j < H_; j += nT) {              // xb_in = Wb[:, :D] @ ctrl + bb
    float a = bb[j];
    for (int k = 0; k < D_; ++k) a = fmaf(Wb[(size_t)j * (D_ + H_) + k], ctrl[k], a);
    xb_in[j] = a;
  }
  for (int i = tid; i < 2 * H_ * D_; i += nT) WinitB[i] = f2bf(Winit[i]);
  for (int i = tid; i < 4 * H_ * H_; i += nT) WrB[i]    = f2bf(Wr[i]);
  for (int i = tid; i < H_ * H_; i += nT) {
    int n = i >> 7, k = i & 127;
    WbhB[i] = f2bf(Wb[(size_t)n * (D_ + H_) + D_ + k]);
    W1B[i]  = f2bf(W1[i]);
    W2B[i]  = f2bf(W2[i]);
    WaB[i]  = f2bf(Wa[i]);
    WtbB[i] = f2bf(Wtb[i]);
  }
}

// ---------------------------------------------------------------------------
// Main fused kernel: 128 agent rows / block, 8 waves, wave w owns output
// column group 16w. c-state stays in registers across all T steps; h-like
// activations ping-pong between two bf16 LDS buffers.
// ---------------------------------------------------------------------------
__global__ __launch_bounds__(256) void cfc_main(
    const float* __restrict__ agent,
    const u16* __restrict__ WinitB, const u16* __restrict__ WrB,
    const u16* __restrict__ WbhB,  const u16* __restrict__ W1B,
    const u16* __restrict__ W2B,   const u16* __restrict__ WaB,
    const u16* __restrict__ WtbB,
    const float* __restrict__ lstm_in, const float* __restrict__ xb_in,
    const float* __restrict__ b_init,
    const float* __restrict__ b1, const float* __restrict__ b2,
    const float* __restrict__ ba, const float* __restrict__ btb,
    const float* __restrict__ Wfc, const float* __restrict__ bfc,
    float* __restrict__ out)
{
  extern __shared__ u16 lds[];
  u16*   bufA = lds;                         // ROWS*HP bf16
  u16*   bufB = lds + ROWS * HP;             // ROWS*HP bf16
  float* sv   = (float*)(lds + 2 * ROWS * HP);

  const int tid  = threadIdx.x;
  const int lane = tid & 31;
  const int wave = tid >> 5;
  const int nb   = 16 * wave;                // this wave's column base
  const int row0 = blockIdx.x * ROWS;

  // stage small vectors into LDS
  for (int i = tid; i < 512; i += 256) sv[SV_LSTM + i] = lstm_in[i];
  for (int i = tid; i < 128; i += 256) {
    sv[SV_XB  + i] = xb_in[i];
    sv[SV_B1  + i] = b1[i];
    sv[SV_B2  + i] = b2[i];
    sv[SV_BAB + i] = ba[i] + btb[i];
  }
  for (int i = tid; i < 256; i += 256) {
    sv[SV_WFC   + i] = Wfc[i];
    sv[SV_BINIT + i] = b_init[i];
  }
  if (tid < 2) sv[SV_BFC + tid] = bfc[tid];
  __syncthreads();

  // ------------------- init GEMM: hc = agent @ Winit^T + b_init -------------
  f32x8 cReg[8];        // c state: row-tile r, cols nb..nb+15 (C layout)
  f32x8 accH[8];
  {
    const float bH = sv[SV_BINIT + nb + (lane & 15)];
    const float bC = sv[SV_BINIT + 128 + nb + (lane & 15)];
#pragma unroll
    for (int r = 0; r < 8; ++r) { accH[r] = splat8(bH); cReg[r] = splat8(bC); }

#pragma unroll 1
    for (int kc = 0; kc < 4; ++kc) {         // K = 512 in 4 chunks of 128
      __syncthreads();                        // protect bufB vs previous readers
      // cooperative stage agent[:, kc*128 .. +128) -> bufB (f32 -> bf16)
      for (int i = tid; i < ROWS * 32; i += 256) {
        int r = i >> 5, c4 = (i & 31) << 2;
        float4 v = *(const float4*)(agent + (size_t)(row0 + r) * D_ + kc * 128 + c4);
        uint2 pk;
        pk.x = (unsigned)f2bf(v.x) | ((unsigned)f2bf(v.y) << 16);
        pk.y = (unsigned)f2bf(v.z) | ((unsigned)f2bf(v.w) << 16);
        *(uint2*)(bufB + r * HP + c4) = pk;
      }
      __syncthreads();
#pragma unroll
      for (int r = 0; r < 8; ++r) {
#pragma unroll
        for (int k8 = 0; k8 < 4; ++k8) {
          bf16x16 a  = ldsA(bufB, 16 * r, 32 * k8, lane);
          bf16x16 bh = gB(WinitB, D_, nb,       kc * 128 + 32 * k8, lane);
          bf16x16 bc = gB(WinitB, D_, 128 + nb, kc * 128 + 32 * k8, lane);
          accH[r] = wmma_bf16(a, bh, accH[r]);
          cReg[r] = wmma_bf16(a, bc, cReg[r]);
        }
      }
    }
#pragma unroll
    for (int r = 0; r < 8; ++r) stC(bufA, 16 * r, nb, lane, accH[r]);  // h0
    __syncthreads();
  }

  // per-lane column biases
  const int   cl  = nb + (lane & 15);
  const float bzi = sv[SV_LSTM + cl];
  const float bzg = sv[SV_LSTM + 128 + cl];
  const float bzf = sv[SV_LSTM + 256 + cl];
  const float bzo = sv[SV_LSTM + 384 + cl];
  const float bx  = sv[SV_XB  + cl];
  const float bb1 = sv[SV_B1  + cl];
  const float bb2 = sv[SV_B2  + cl];
  const float bbt = sv[SV_BAB + cl];

  u16* src = bufA;                           // holds h carry
  u16* dst = bufB;

#pragma unroll 1
  for (int t = 0; t < T_; ++t) {
    // ---- P1: z = lstm_in + h @ Wr^T ; LSTM gates; h_lstm -> dst ----
#pragma unroll
    for (int r = 0; r < 8; ++r) {
      f32x8 zi = splat8(bzi), zg = splat8(bzg), zf = splat8(bzf), zo = splat8(bzo);
#pragma unroll
      for (int k8 = 0; k8 < 4; ++k8) {
        bf16x16 a = ldsA(src, 16 * r, 32 * k8, lane);
        zi = wmma_bf16(a, gB(WrB, H_, nb,       32 * k8, lane), zi);
        zg = wmma_bf16(a, gB(WrB, H_, 128 + nb, 32 * k8, lane), zg);
        zf = wmma_bf16(a, gB(WrB, H_, 256 + nb, 32 * k8, lane), zf);
        zo = wmma_bf16(a, gB(WrB, H_, 384 + nb, 32 * k8, lane), zo);
      }
      f32x8 hl;
#pragma unroll
      for (int e = 0; e < 8; ++e) {
        float cn = cReg[r][e] * sigm(zf[e] + 1.0f) + tanh_fast(zi[e]) * sigm(zg[e]);
        cReg[r][e] = cn;
        hl[e] = tanh_fast(cn) * sigm(zo[e]);
      }
      stC(dst, 16 * r, nb, lane, hl);
    }
    __syncthreads();

    // ---- P2: x = lecun_tanh(xb_in + h_lstm @ Wbh^T) ; x -> src ----
#pragma unroll
    for (int r = 0; r < 8; ++r) {
      f32x8 xa = splat8(bx);
#pragma unroll
      for (int k8 = 0; k8 < 4; ++k8) {
        bf16x16 a = ldsA(dst, 16 * r, 32 * k8, lane);
        xa = wmma_bf16(a, gB(WbhB, H_, nb, 32 * k8, lane), xa);
      }
      f32x8 xv;
#pragma unroll
      for (int e = 0; e < 8; ++e) xv[e] = 1.7159f * tanh_fast(0.666f * xa[e]);
      stC(src, 16 * r, nb, lane, xv);
    }
    __syncthreads();

    // ---- P3: ff1/ff2/t_interp ; h_new -> dst ----
#pragma unroll
    for (int r = 0; r < 8; ++r) {
      f32x8 f1 = splat8(bb1), f2v = splat8(bb2), ts = splat8(bbt);
#pragma unroll
      for (int k8 = 0; k8 < 4; ++k8) {
        bf16x16 a = ldsA(src, 16 * r, 32 * k8, lane);
        f1  = wmma_bf16(a, gB(W1B,  H_, nb, 32 * k8, lane), f1);
        f2v = wmma_bf16(a, gB(W2B,  H_, nb, 32 * k8, lane), f2v);
        ts  = wmma_bf16(a, gB(WaB,  H_, nb, 32 * k8, lane), ts);
        ts  = wmma_bf16(a, gB(WtbB, H_, nb, 32 * k8, lane), ts);
      }
      f32x8 hn;
#pragma unroll
      for (int e = 0; e < 8; ++e) {
        float ti = sigm(ts[e]);
        hn[e] = tanh_fast(f1[e]) * (1.0f - ti) + ti * tanh_fast(f2v[e]);
      }
      stC(dst, 16 * r, nb, lane, hn);
    }
    __syncthreads();

    // ---- P4: y = h_new @ Wfc^T + bfc (read-only on dst; no barrier needed) --
    {
      int rr = tid >> 1, o = tid & 1;
      const u16*   hp = dst + rr * HP;
      const float* wp = sv + SV_WFC + o * 128;
      float acc = sv[SV_BFC + o];
#pragma unroll 8
      for (int k = 0; k < 128; ++k) acc = fmaf(bf2f(hp[k]), wp[k], acc);
      out[(size_t)(row0 + rr) * (T_ * 2) + t * 2 + o] = acc;
    }

    u16* tmp = src; src = dst; dst = tmp;    // h carry = h_new
  }
}

// ---------------------------------------------------------------------------
extern "C" void kernel_launch(void* const* d_in, const int* in_sizes, int n_in,
                              void* d_out, int out_size, void* d_ws, size_t ws_size,
                              hipStream_t stream)
{
  (void)n_in; (void)out_size; (void)ws_size;
  const float* agent = (const float*)d_in[0];
  const float* ctrl  = (const float*)d_in[1];
  const float* Winit = (const float*)d_in[2];
  const float* binit = (const float*)d_in[3];
  const float* Wi    = (const float*)d_in[4];
  const float* bi    = (const float*)d_in[5];
  const float* Wr    = (const float*)d_in[6];
  const float* Wb    = (const float*)d_in[7];
  const float* bb    = (const float*)d_in[8];
  const float* W1    = (const float*)d_in[9];
  const float* b1    = (const float*)d_in[10];
  const float* W2    = (const float*)d_in[11];
  const float* b2    = (const float*)d_in[12];
  const float* Wa    = (const float*)d_in[13];
  const float* ba    = (const float*)d_in[14];
  const float* Wtb   = (const float*)d_in[15];
  const float* btb   = (const float*)d_in[16];
  const float* Wfc   = (const float*)d_in[17];
  const float* bfc   = (const float*)d_in[18];

  char* ws = (char*)d_ws;
  u16*   WinitB = (u16*)(ws + 0);        // 256*512*2 = 262144
  u16*   WrB    = (u16*)(ws + 262144);   // 512*128*2 = 131072
  u16*   WbhB   = (u16*)(ws + 393216);   // 128*128*2 =  32768
  u16*   W1B    = (u16*)(ws + 425984);
  u16*   W2B    = (u16*)(ws + 458752);
  u16*   WaB    = (u16*)(ws + 491520);
  u16*   WtbB   = (u16*)(ws + 524288);
  float* lstm   = (float*)(ws + 557056); // 512*4
  float* xb     = (float*)(ws + 559104); // 128*4

  cfc_prep<<<64, 256, 0, stream>>>(ctrl, Winit, Wi, bi, Wr, Wb, bb, W1, W2, Wa, Wtb,
                                   WinitB, WrB, WbhB, W1B, W2B, WaB, WtbB, lstm, xb);

  int n       = in_sizes[0] / D_;
  int nblocks = n / ROWS;
  size_t ldsBytes = (size_t)2 * ROWS * HP * sizeof(u16) + SV_FLOATS * sizeof(float);
  cfc_main<<<nblocks, 256, ldsBytes, stream>>>(agent, WinitB, WrB, WbhB, W1B, W2B, WaB, WtbB,
                                               lstm, xb, binit, b1, b2, ba, btb, Wfc, bfc,
                                               (float*)d_out);
}
